// InnerBlock_32083405701821
// MI455X (gfx1250) — matrix operand order repaired and checked by reading, not compiled, not run on hardware
//
#include <hip/hip_runtime.h>

#define DEV __device__ __forceinline__

typedef __attribute__((ext_vector_type(16))) __bf16 v16bf;
typedef __attribute__((ext_vector_type(8)))  float  v8f;

// ---------------- model dimensions ----------------
constexpr int B_   = 8;
constexpr int C_   = 128;
constexpr int WC_  = 64;
constexpr int HD_  = 16;
constexpr int AGC_ = 32;
constexpr int HS_  = 64;        // x spatial
constexpr int HW_  = HS_ * HS_; // 4096
constexpr int IS_  = 256;       // image spatial
constexpr int HID_ = 341;       // int(8/3*128)
constexpr int N1_  = 2 * HID_;  // 682
constexpr int N1P_ = 688;       // 682 padded to /16
constexpr int K2P_ = 352;       // 341 padded to /32
constexpr float ATT_SCALE_ = 0.11180339887498949f; // (0.5*(C+AGC))^-0.5

// ---------------- bf16 helpers ----------------
DEV __bf16 bfz() { return __builtin_bit_cast(__bf16, (unsigned short)0); }
DEV __bf16 f2bf(float f) {
  unsigned u = __builtin_bit_cast(unsigned, f);
  unsigned r = u + 0x7FFFu + ((u >> 16) & 1u);
  unsigned short h = (unsigned short)(r >> 16);
  return __builtin_bit_cast(__bf16, h);
}

// ============ 1. RGB -> HSL ============
__global__ void k_rgb2hsl(const float* __restrict__ img, float* __restrict__ hsl, int npix) {
  int p = blockIdx.x * blockDim.x + threadIdx.x;
  if (p >= npix) return;
  int b = p / (IS_ * IS_);
  int s = p % (IS_ * IS_);
  const float* base = img + (size_t)b * 3 * IS_ * IS_ + s;
  float r = base[0], g = base[(size_t)IS_ * IS_], bl = base[(size_t)2 * IS_ * IS_];
  float cmax = fmaxf(r, fmaxf(g, bl));
  float cmin = fminf(r, fminf(g, bl));
  float delta = cmax - cmin;
  int idx = (r >= g && r >= bl) ? 0 : ((g >= bl) ? 1 : 2);
  float sd = (delta == 0.f) ? 1.f : delta;
  float h;
  if (delta == 0.f) h = 0.f;
  else if (idx == 0) { h = fmodf((g - bl) / sd, 6.f); if (h < 0.f) h += 6.f; }
  else if (idx == 1) h = (bl - r) / sd + 2.f;
  else               h = (r - g) / sd + 4.f;
  h *= (1.f / 6.f);
  float l = 0.5f * (cmax + cmin);
  float den = (l <= 0.5f) ? 2.f * l : 2.f - 2.f * l;
  float ss = ((l > 0.f) && (l < 1.f)) ? delta / ((den == 0.f) ? 1.f : den) : 0.f;
  float* ob = hsl + (size_t)b * 3 * IS_ * IS_ + s;
  ob[0] = h; ob[(size_t)IS_ * IS_] = ss; ob[(size_t)2 * IS_ * IS_] = l;
}

// ============ 2. 5x5 conv, pad 2, stride 1, 3 out channels ============
__global__ void k_conv5(const float* __restrict__ hsl, int choff, int Cin,
                        const float* __restrict__ w, const float* __restrict__ bias,
                        float* __restrict__ out) {
  size_t n = (size_t)B_ * 3 * IS_ * IS_;
  size_t t = (size_t)blockIdx.x * blockDim.x + threadIdx.x;
  if (t >= n) return;
  int x = t % IS_;
  int y = (t / IS_) % IS_;
  int oc = (t / ((size_t)IS_ * IS_)) % 3;
  int b = t / ((size_t)3 * IS_ * IS_);
  float acc = bias[oc];
  for (int ic = 0; ic < Cin; ++ic)
    for (int ky = 0; ky < 5; ++ky) {
      int iy = y + ky - 2; if (iy < 0 || iy >= IS_) continue;
      for (int kx = 0; kx < 5; ++kx) {
        int ix = x + kx - 2; if (ix < 0 || ix >= IS_) continue;
        acc += hsl[(((size_t)b * 3 + choff + ic) * IS_ + iy) * IS_ + ix] *
               w[((oc * Cin + ic) * 5 + ky) * 5 + kx];
      }
    }
  out[t] = acc;
}

// ============ 3. distill: 2x2 stride2 conv + leaky relu ============
__global__ void k_dist_conv(const float* __restrict__ in, int Cin, int Si,
                            const float* __restrict__ w, const float* __restrict__ bias,
                            const float* __restrict__ aslope, int Cout,
                            float* __restrict__ out) {
  int So = Si >> 1;
  size_t n = (size_t)B_ * Cout * So * So;
  size_t t = (size_t)blockIdx.x * blockDim.x + threadIdx.x;
  if (t >= n) return;
  int x = t % So;
  int y = (t / So) % So;
  int oc = (t / ((size_t)So * So)) % Cout;
  int b = t / ((size_t)Cout * So * So);
  float acc = bias[oc];
  for (int ic = 0; ic < Cin; ++ic)
    for (int ky = 0; ky < 2; ++ky)
      for (int kx = 0; kx < 2; ++kx)
        acc += in[(((size_t)b * Cin + ic) * Si + (2 * y + ky)) * Si + (2 * x + kx)] *
               w[((oc * Cin + ic) * 2 + ky) * 2 + kx];
  float a = aslope[0];
  out[t] = (acc >= 0.f) ? acc : a * acc;
}

// per-channel mean/var over (B, H, W)
__global__ void k_chstats(const float* __restrict__ x, int Cc, int SS,
                          float* __restrict__ mean, float* __restrict__ var) {
  int ch = blockIdx.x;
  __shared__ float s1[256], s2[256];
  float a1 = 0.f, a2 = 0.f;
  int n = B_ * SS;
  for (int i = threadIdx.x; i < n; i += 256) {
    int b = i / SS, s = i % SS;
    float v = x[((size_t)b * Cc + ch) * SS + s];
    a1 += v; a2 += v * v;
  }
  s1[threadIdx.x] = a1; s2[threadIdx.x] = a2; __syncthreads();
  for (int st = 128; st > 0; st >>= 1) {
    if (threadIdx.x < st) { s1[threadIdx.x] += s1[threadIdx.x + st]; s2[threadIdx.x] += s2[threadIdx.x + st]; }
    __syncthreads();
  }
  if (threadIdx.x == 0) {
    float m = s1[0] / n;
    mean[ch] = m;
    var[ch] = s2[0] / n - m * m;
  }
}

__global__ void k_dist_norm(float* __restrict__ x, int Cc, int SS,
                            const float* __restrict__ mean, const float* __restrict__ var,
                            const float* __restrict__ g, const float* __restrict__ bt) {
  size_t n = (size_t)B_ * Cc * SS;
  size_t t = (size_t)blockIdx.x * blockDim.x + threadIdx.x;
  if (t >= n) return;
  int ch = (t / SS) % Cc;
  x[t] = (x[t] - mean[ch]) * rsqrtf(var[ch] + 1e-5f) * g[ch] + bt[ch];
}

// ============ 4. channel LayerNorm (per pixel over C), optional bf16 channel-last ============
__global__ void k_ln_ch(const float* __restrict__ x, const float* __restrict__ g,
                        const float* __restrict__ bb, int Cc, float eps,
                        float* __restrict__ out, __bf16* __restrict__ out_cl) {
  int p = blockIdx.x * blockDim.x + threadIdx.x;
  if (p >= B_ * HW_) return;
  int b = p / HW_, s = p % HW_;
  const float* xb = x + (size_t)b * Cc * HW_ + s;
  float mu = 0.f;
  for (int c = 0; c < Cc; ++c) mu += xb[(size_t)c * HW_];
  mu /= Cc;
  float var = 0.f;
  for (int c = 0; c < Cc; ++c) { float d = xb[(size_t)c * HW_] - mu; var += d * d; }
  var /= Cc;
  float rs = rsqrtf(var + eps);
  for (int c = 0; c < Cc; ++c) {
    float v = (xb[(size_t)c * HW_] - mu) * rs * g[c] + bb[c];
    if (out)    out[((size_t)b * Cc + c) * HW_ + s] = v;
    if (out_cl) out_cl[(size_t)p * Cc + c] = f2bf(v);
  }
}

// ============ 5. 1x1 conv -> channel-last f32 (supports 2-input concat), Cout = HD ============
__global__ void k_conv1x1_cl(const float* __restrict__ in1, int C1,
                             const float* __restrict__ in2, int C2,
                             const float* __restrict__ w, const float* __restrict__ bias,
                             float* __restrict__ out_cl) {
  int p = blockIdx.x * blockDim.x + threadIdx.x;
  if (p >= B_ * HW_) return;
  int b = p / HW_, s = p % HW_;
  int CT = C1 + C2;
  float acc[HD_];
#pragma unroll
  for (int o = 0; o < HD_; ++o) acc[o] = bias[o];
  for (int c = 0; c < C1; ++c) {
    float v = in1[((size_t)b * C1 + c) * HW_ + s];
#pragma unroll
    for (int o = 0; o < HD_; ++o) acc[o] += w[o * CT + c] * v;
  }
  for (int c = 0; c < C2; ++c) {
    float v = in2[((size_t)b * C2 + c) * HW_ + s];
#pragma unroll
    for (int o = 0; o < HD_; ++o) acc[o] += w[o * CT + C1 + c] * v;
  }
#pragma unroll
  for (int o = 0; o < HD_; ++o) out_cl[(size_t)p * HD_ + o] = acc[o];
}

// mean over pixels of channel-last (B*HW, HD) -> (B, HD)
__global__ void k_mean_cl(const float* __restrict__ xcl, float* __restrict__ out) {
  int b = blockIdx.x / HD_, o = blockIdx.x % HD_;
  __shared__ float sm[256];
  float a = 0.f;
  for (int p = threadIdx.x; p < HW_; p += 256)
    a += xcl[((size_t)b * HW_ + p) * HD_ + o];
  sm[threadIdx.x] = a; __syncthreads();
  for (int st = 128; st > 0; st >>= 1) {
    if (threadIdx.x < st) sm[threadIdx.x] += sm[threadIdx.x + st];
    __syncthreads();
  }
  if (threadIdx.x == 0) out[b * HD_ + o] = sm[0] / HW_;
}

// alpha[b] = ap_w . (ca_w @ mean_k + ca_b);  omega likewise (conv+mean commute)
__global__ void k_alpha(const float* __restrict__ kmean, const float* __restrict__ qmean,
                        const float* __restrict__ caw, const float* __restrict__ cab,
                        const float* __restrict__ apw,
                        const float* __restrict__ cow, const float* __restrict__ cob,
                        const float* __restrict__ opw,
                        float* __restrict__ alpha, float* __restrict__ omega) {
  int b = threadIdx.x;
  if (b < B_) {
    float sa = 0.f, so = 0.f;
    for (int o = 0; o < HD_; ++o) {
      float xa = cab[o], xo = cob[o];
      for (int c = 0; c < HD_; ++c) {
        xa += caw[o * HD_ + c] * kmean[b * HD_ + c];
        xo += cow[o * HD_ + c] * qmean[b * HD_ + c];
      }
      sa += apw[o] * xa;
      so += opw[o] * xo;
    }
    alpha[b] = sa; omega[b] = so;
  }
}

// ============ 6. fused attention per (b, h-row) ============
// WMMA tile of X(64x16) @ Y(64x16)^T, K=16 zero-padded to 32.
// A frag (16x32 bf16): lane<16 elems 0..7 = X[m][0..7] (K 0-7), 8..15 = 0 (K 16-23);
//                      lane>=16 elems 0..7 = X[m][8..15] (K 8-15), 8..15 = 0 (K 24-31).
// B frag: lane<16 elems 0..7 = Y[n][0..7] (K 0-7), lane>=16 elems 0..7 = Y[n][8..15] (K 8-15); elems 8..15 = 0.
DEV void wmma_xyT16(const __bf16* X, const __bf16* Y, int m0, int n0,
                    float scale, float* L, int lane) {
  const __bf16 z = bfz();
  v16bf af, bfr;
#pragma unroll
  for (int i = 0; i < 16; ++i) { af[i] = z; bfr[i] = z; }
  int half = lane >> 4;
  int m = m0 + (lane & 15);
  const __bf16* xr = X + m * 16 + (half ? 8 : 0);
#pragma unroll
  for (int i = 0; i < 8; ++i) af[i] = xr[i];
  {
    const __bf16* yr = Y + (n0 + (lane & 15)) * 16 + (half ? 8 : 0);
#pragma unroll
    for (int i = 0; i < 8; ++i) bfr[i] = yr[i];
  }
  v8f c = {};
  c = __builtin_amdgcn_wmma_f32_16x16x32_bf16(false, af, false, bfr, (short)0, c, false, false);
  int n = n0 + (lane & 15);
  int mb = m0 + (half ? 8 : 0);
#pragma unroll
  for (int r = 0; r < 8; ++r) L[(mb + r) * 64 + n] = scale * c[r];
}

__global__ void k_attention(const float* __restrict__ kcl, const float* __restrict__ qcl,
                            const float* __restrict__ acl, const float* __restrict__ vcl,
                            const float* __restrict__ alphaArr, const float* __restrict__ omegaArr,
                            float* __restrict__ mp) {
  __shared__ __bf16 Kt[64 * 16], At[64 * 16], Qt[64 * 16], Vt[64 * 16];
  __shared__ float L1[64 * 64], L2[64 * 64];
  __shared__ __bf16 Sb[64 * 64];
  const int b = blockIdx.x >> 6;
  const int h = blockIdx.x & 63;
  const int t = threadIdx.x;
  const size_t rowbase = ((size_t)b * HW_ + (size_t)h * 64) * HD_;
  for (int i = t; i < 64 * 16; i += 256) {
    Kt[i] = f2bf(kcl[rowbase + i]);
    At[i] = f2bf(acl[rowbase + i]);
    Qt[i] = f2bf(qcl[rowbase + i]);
    Vt[i] = f2bf(vcl[rowbase + i]);
  }
  __syncthreads();
  const int wv = t >> 5, lane = t & 31;
  // 32 logit tiles (16 for L1 = K@A^T, 16 for L2 = A@Q^T) over 8 waves
  for (int tt = wv * 4; tt < wv * 4 + 4; ++tt) {
    int which = tt >> 4;
    int sub = tt & 15;
    int m0 = (sub >> 2) * 16, n0 = (sub & 3) * 16;
    wmma_xyT16(which ? At : Kt, which ? Qt : At, m0, n0, ATT_SCALE_,
               which ? L2 : L1, lane);
  }
  __syncthreads();
  // row softmax (axis = v)
  if (t < 128) {
    float* row = (t < 64) ? (L1 + t * 64) : (L2 + (t - 64) * 64);
    float mx = row[0];
    for (int i = 1; i < 64; ++i) mx = fmaxf(mx, row[i]);
    float s = 0.f;
    for (int i = 0; i < 64; ++i) { float e = expf(row[i] - mx); row[i] = e; s += e; }
    float inv = 1.f / s;
    for (int i = 0; i < 64; ++i) row[i] *= inv;
  }
  __syncthreads();
  const float al = alphaArr[b], om = omegaArr[b];
  if (t < 64) {
    for (int v = 0; v < 64; ++v)
      Sb[t * 64 + v] = f2bf(al * L1[t * 64 + v] + om * L2[t * 64 + v]);
  }
  __syncthreads();
  // mp = S(64x64) @ V(64x16); 4 m-tiles on waves 0..3, K-loop of 2 WMMAs
  if (wv < 4) {
    int m0 = wv * 16;
    v8f acc = {};
    int half = lane >> 4;
    int mrow = m0 + (lane & 15);
    int n = lane & 15;
    for (int k0 = 0; k0 < 64; k0 += 32) {
      v16bf af, bfr;
      int ka = k0 + (half ? 8 : 0);
#pragma unroll
      for (int i = 0; i < 8; ++i) {
        af[i]     = Sb[mrow * 64 + ka + i];
        af[i + 8] = Sb[mrow * 64 + ka + 16 + i];
      }
      int kb = k0 + (half ? 8 : 0);
#pragma unroll
      for (int i = 0; i < 8; ++i) { bfr[i] = Vt[(kb + i) * 16 + n]; bfr[i + 8] = Vt[(kb + 16 + i) * 16 + n]; }
      acc = __builtin_amdgcn_wmma_f32_16x16x32_bf16(false, af, false, bfr, (short)0, acc, false, false);
    }
    int mb = m0 + (half ? 8 : 0);
#pragma unroll
    for (int r = 0; r < 8; ++r)
      mp[((size_t)b * HW_ + (size_t)h * 64 + (mb + r)) * HD_ + n] = acc[r];
  }
}

// ============ 7. attention epilogue: gamma*(fan 1x1) + dwconv3x3(xn) ============
__global__ void k_attn_out(const float* __restrict__ mp, const float* __restrict__ xn,
                           const float* __restrict__ fanw, const float* __restrict__ fanb,
                           const float* __restrict__ gamma,
                           const float* __restrict__ dww, const float* __restrict__ dwb,
                           float* __restrict__ xla) {
  size_t t = (size_t)blockIdx.x * blockDim.x + threadIdx.x;
  if (t >= (size_t)B_ * C_ * HW_) return;
  int s = t % HW_;
  int cc = (t / HW_) % C_;
  int b = t / ((size_t)C_ * HW_);
  int y = s >> 6, x = s & 63;
  const float* mprow = mp + ((size_t)b * HW_ + s) * HD_;
  float fan = fanb[cc];
#pragma unroll
  for (int j = 0; j < HD_; ++j) fan += fanw[cc * HD_ + j] * mprow[j];
  float dw = dwb[cc];
  const float* xc = xn + ((size_t)b * C_ + cc) * HW_;
  for (int ky = 0; ky < 3; ++ky) {
    int iy = y + ky - 1; if (iy < 0 || iy >= HS_) continue;
    for (int kx = 0; kx < 3; ++kx) {
      int ix = x + kx - 1; if (ix < 0 || ix >= HS_) continue;
      dw += dww[cc * 9 + ky * 3 + kx] * xc[iy * HS_ + ix];
    }
  }
  xla[t] = gamma[cc] * fan + dw;
}

// ============ 8. weight convert to padded bf16 (rows x cols -> prows x pcols) ============
__global__ void k_wcvt(const float* __restrict__ w, int rows, int cols, int prows, int pcols,
                       __bf16* __restrict__ out) {
  size_t n = (size_t)prows * pcols;
  size_t t = (size_t)blockIdx.x * blockDim.x + threadIdx.x;
  if (t >= n) return;
  int r = t / pcols, c = t % pcols;
  out[t] = (r < rows && c < cols) ? f2bf(w[(size_t)r * cols + c]) : bfz();
}

// ============ 9. bf16 WMMA GEMM: C(MxNpad) = A(MxKa) * W(NxKa)^T + bias ============
// One wave -> one 16x16 tile. Fragment layouts as in wmma_xyT16 (full K here).
__global__ void k_gemm(const __bf16* __restrict__ A, const __bf16* __restrict__ W,
                       const float* __restrict__ bias, int Ka, int Npad, int Nreal,
                       int totalTiles, float* __restrict__ Cmat) {
  int wv = threadIdx.x >> 5, lane = threadIdx.x & 31;
  int tile = blockIdx.x * 4 + wv;
  if (tile >= totalTiles) return;
  int NT = Npad >> 4;
  int mt = tile / NT, nt = tile % NT;
  int m = mt * 16 + (lane & 15);
  int n = nt * 16 + (lane & 15);
  int half = lane >> 4;
  v8f acc = {};
  for (int k0 = 0; k0 < Ka; k0 += 32) {
    v16bf af, bfr;
    const __bf16* ap = A + (size_t)m * Ka + k0 + (half ? 8 : 0);
#pragma unroll
    for (int i = 0; i < 8; ++i) { af[i] = ap[i]; af[i + 8] = ap[16 + i]; }
    const __bf16* bp = W + (size_t)n * Ka + k0 + (half ? 8 : 0);
#pragma unroll
    for (int i = 0; i < 8; ++i) { bfr[i] = bp[i]; bfr[i + 8] = bp[16 + i]; }
    acc = __builtin_amdgcn_wmma_f32_16x16x32_bf16(false, af, false, bfr, (short)0, acc, false, false);
  }
  float bn = (n < Nreal) ? bias[n] : 0.f;
  int mb = mt * 16 + (half ? 8 : 0);
#pragma unroll
  for (int r = 0; r < 8; ++r)
    Cmat[(size_t)(mb + r) * Npad + n] = acc[r] + bn;
}

// ============ 10. gated-CNN plumbing ============
__global__ void k_extract_c(const float* __restrict__ y, float* __restrict__ cbuf) {
  size_t n = (size_t)B_ * C_ * HW_;
  size_t t = (size_t)blockIdx.x * blockDim.x + threadIdx.x;
  if (t >= n) return;
  int s = t % HW_;
  int cc = (t / HW_) % C_;
  int b = t / ((size_t)C_ * HW_);
  cbuf[t] = y[((size_t)b * HW_ + s) * N1P_ + (N1_ - C_) + cc]; // cols 554..681
}

__global__ void k_dwconv7(const float* __restrict__ in, const float* __restrict__ w,
                          const float* __restrict__ bias, float* __restrict__ out) {
  size_t n = (size_t)B_ * C_ * HW_;
  size_t t = (size_t)blockIdx.x * blockDim.x + threadIdx.x;
  if (t >= n) return;
  int s = t % HW_;
  int cc = (t / HW_) % C_;
  int b = t / ((size_t)C_ * HW_);
  int y = s >> 6, x = s & 63;
  float acc = bias[cc];
  for (int ky = 0; ky < 7; ++ky) {
    int iy = y + ky - 3; if (iy < 0 || iy >= HS_) continue;
    for (int kx = 0; kx < 7; ++kx) {
      int ix = x + kx - 3; if (ix < 0 || ix >= HS_) continue;
      acc += w[cc * 49 + ky * 7 + kx] * in[((size_t)b * C_ + cc) * HW_ + iy * HS_ + ix];
    }
  }
  out[t] = acc;
}

__global__ void k_buildU(const float* __restrict__ y, const float* __restrict__ cconv,
                         __bf16* __restrict__ U) {
  size_t n = (size_t)B_ * HW_ * K2P_;
  size_t t = (size_t)blockIdx.x * blockDim.x + threadIdx.x;
  if (t >= n) return;
  int j = t % K2P_;
  size_t p = t / K2P_;
  if (j >= HID_) { U[t] = bfz(); return; }
  float gv = y[p * N1P_ + j];
  float gel = 0.5f * gv * (1.f + erff(gv * 0.70710678118654752f));
  float other;
  if (j < HID_ - C_) other = y[p * N1P_ + HID_ + j];
  else {
    int cc = j - (HID_ - C_);
    int b = p / HW_, s = p % HW_;
    other = cconv[((size_t)b * C_ + cc) * HW_ + s];
  }
  U[t] = f2bf(gel * other);
}

__global__ void k_final_l(const float* __restrict__ xl, const float* __restrict__ xla,
                          const float* __restrict__ go, const float* __restrict__ alphal,
                          float* __restrict__ out) {
  size_t n = (size_t)B_ * C_ * HW_;
  size_t t = (size_t)blockIdx.x * blockDim.x + threadIdx.x;
  if (t >= n) return;
  int s = t % HW_;
  int cc = (t / HW_) % C_;
  int b = t / ((size_t)C_ * HW_);
  size_t p = (size_t)b * HW_ + s;
  out[t] = alphal[0] * xl[t] + xla[t] + go[p * C_ + cc];
}

// ============ 11. SE path ============
__global__ void k_se_reduce(const float* __restrict__ x, float* __restrict__ ymean) {
  int b = blockIdx.x / C_, cc = blockIdx.x % C_;
  __shared__ float sm[256];
  float a = 0.f;
  for (int s = threadIdx.x; s < HW_; s += 256)
    a += x[((size_t)b * C_ + cc) * HW_ + s];
  sm[threadIdx.x] = a; __syncthreads();
  for (int st = 128; st > 0; st >>= 1) {
    if (threadIdx.x < st) sm[threadIdx.x] += sm[threadIdx.x + st];
    __syncthreads();
  }
  if (threadIdx.x == 0) ymean[b * C_ + cc] = sm[0] / HW_;
}

__global__ void k_se_gate(const float* __restrict__ ymean, const float* __restrict__ w1,
                          const float* __restrict__ w2, float* __restrict__ yg) {
  int b = blockIdx.x;
  __shared__ float hid[8];
  if (threadIdx.x < 8) {
    float a = 0.f;
    for (int c = 0; c < C_; ++c) a += w1[threadIdx.x * C_ + c] * ymean[b * C_ + c];
    hid[threadIdx.x] = fmaxf(a, 0.f);
  }
  __syncthreads();
  if (threadIdx.x < C_) {
    float a = 0.f;
    for (int j = 0; j < 8; ++j) a += w2[threadIdx.x * 8 + j] * hid[j];
    yg[b * C_ + threadIdx.x] = 1.f / (1.f + expf(-a));
  }
}

__global__ void k_final_r(const float* __restrict__ xr, const float* __restrict__ xra,
                          const float* __restrict__ yg, const float* __restrict__ alphar,
                          float* __restrict__ out) {
  size_t n = (size_t)B_ * C_ * HW_;
  size_t t = (size_t)blockIdx.x * blockDim.x + threadIdx.x;
  if (t >= n) return;
  int cc = (t / HW_) % C_;
  int b = t / ((size_t)C_ * HW_);
  out[t] = alphar[0] * xr[t] + xra[t] * yg[b * C_ + cc];
}

// ====================================================================
// Host orchestration.
// Input index map (jax sorted-key pytree flatten of setup_inputs()):
// 0 xl, 1 wl_l, 2 xr, 3 wr_l, 4 wl_h, 5 wr_h, 6 i,
// 7 alpha_l, 8 alpha_r,
// ca_l @9, ca_r @34  (25 leaves each, sorted: ag_b ag_w ap_w ca_b ca_w co_b co_w
//   dw_b dw_w fan_b fan_w gamma k_b k_w ln1_b ln1_g ln2_b ln2_g ln3_b ln3_g
//   op_w q_b q_w v_b v_w)
// dist_l @59, dist_r @69 (per stage: a b bt g w)
// gcnn @79: conv_b conv_w fc1_b fc1_w fc2_b fc2_w ln_b ln_g
// se  @87: w1 w2
// split @89: b1 b2 w1 w2
// ====================================================================
enum { A_AG_B = 0, A_AG_W, A_AP_W, A_CA_B, A_CA_W, A_CO_B, A_CO_W, A_DW_B, A_DW_W,
       A_FAN_B, A_FAN_W, A_GAMMA, A_K_B, A_K_W, A_LN1_B, A_LN1_G, A_LN2_B, A_LN2_G,
       A_LN3_B, A_LN3_G, A_OP_W, A_Q_B, A_Q_W, A_V_B, A_V_W };

extern "C" void kernel_launch(void* const* d_in, const int* in_sizes, int n_in,
                              void* d_out, int out_size, void* d_ws, size_t ws_size,
                              hipStream_t stream) {
  (void)in_sizes; (void)n_in; (void)out_size; (void)ws_size;
  auto F = [&](int i) { return (const float*)d_in[i]; };

  char* wsb = (char*)d_ws;
  size_t off = 0;
  auto alloc = [&](size_t bytes) {
    void* p = wsb + off;
    off += (bytes + 255) & ~(size_t)255;
    return p;
  };

  float* HSLb  = (float*)alloc((size_t)B_ * 3 * IS_ * IS_ * 4);
  float* SLb   = (float*)alloc((size_t)B_ * 3 * IS_ * IS_ * 4);
  float* SRb   = (float*)alloc((size_t)B_ * 3 * IS_ * IS_ * 4);
  float* D0b   = (float*)alloc((size_t)B_ * 16 * 128 * 128 * 4);
  float* MEANb = (float*)alloc(64 * 4);
  float* VARb  = (float*)alloc(64 * 4);
  float* AGb   = (float*)alloc((size_t)B_ * AGC_ * HW_ * 4);
  float* XNb   = (float*)alloc((size_t)B_ * C_ * HW_ * 4);
  float* WNb   = (float*)alloc((size_t)B_ * WC_ * HW_ * 4);
  float* AGNb  = (float*)alloc((size_t)B_ * AGC_ * HW_ * 4);
  float* KCL   = (float*)alloc((size_t)B_ * HW_ * HD_ * 4);
  float* QCL   = (float*)alloc((size_t)B_ * HW_ * HD_ * 4);
  float* ACL   = (float*)alloc((size_t)B_ * HW_ * HD_ * 4);
  float* VCL   = (float*)alloc((size_t)B_ * HW_ * HD_ * 4);
  float* KM    = (float*)alloc(B_ * HD_ * 4);
  float* QM    = (float*)alloc(B_ * HD_ * 4);
  float* ALv   = (float*)alloc(64);
  float* OMv   = (float*)alloc(64);
  float* MPb   = (float*)alloc((size_t)B_ * HW_ * HD_ * 4);
  float* XLAb  = (float*)alloc((size_t)B_ * C_ * HW_ * 4);
  __bf16* XLNBF = (__bf16*)alloc((size_t)B_ * HW_ * C_ * 2);
  __bf16* W1BF  = (__bf16*)alloc((size_t)N1P_ * C_ * 2);
  __bf16* W2BF  = (__bf16*)alloc((size_t)C_ * K2P_ * 2);
  float* Yb    = (float*)alloc((size_t)B_ * HW_ * N1P_ * 4);
  float* CBUFb = (float*)alloc((size_t)B_ * C_ * HW_ * 4);
  float* CCONVb= (float*)alloc((size_t)B_ * C_ * HW_ * 4);
  __bf16* UBF  = (__bf16*)alloc((size_t)B_ * HW_ * K2P_ * 2);
  float* GOb   = (float*)alloc((size_t)B_ * HW_ * C_ * 4);
  float* YMEANb= (float*)alloc(B_ * C_ * 4);
  float* YGb   = (float*)alloc(B_ * C_ * 4);

  const int npix = B_ * IS_ * IS_;
  k_rgb2hsl<<<(npix + 255) / 256, 256, 0, stream>>>(F(6), HSLb, npix);

  const size_t nsp = (size_t)B_ * 3 * IS_ * IS_;
  // split: xl from L channel (idx 2, Cin=1) w1/b1;  xr from H,S (idx 0, Cin=2) w2/b2
  k_conv5<<<(int)((nsp + 255) / 256), 256, 0, stream>>>(HSLb, 2, 1, F(91), F(89), SLb);
  k_conv5<<<(int)((nsp + 255) / 256), 256, 0, stream>>>(HSLb, 0, 2, F(92), F(90), SRb);

  const int BCHW = B_ * C_ * HW_;               // 4194304
  const int gBCHW = BCHW / 256;                 // 16384
  const int gPix = (B_ * HW_) / 256;            // 128

  for (int side = 0; side < 2; ++side) {
    const float* xin  = F(side ? 2 : 0);
    const float* win  = F(side ? 3 : 1);
    const float* spl  = side ? SRb : SLb;
    const int D = side ? 69 : 59;
    const int A = side ? 34 : 9;

    // ---- distill: stage0 3->16 (256->128), stage1 16->32 (128->64) ----
    {
      size_t n0 = (size_t)B_ * 16 * 128 * 128;
      k_dist_conv<<<(int)((n0 + 255) / 256), 256, 0, stream>>>(spl, 3, 256, F(D + 4), F(D + 1), F(D + 0), 16, D0b);
      k_chstats<<<16, 256, 0, stream>>>(D0b, 16, 128 * 128, MEANb, VARb);
      k_dist_norm<<<(int)((n0 + 255) / 256), 256, 0, stream>>>(D0b, 16, 128 * 128, MEANb, VARb, F(D + 3), F(D + 2));
      size_t n1 = (size_t)B_ * 32 * HW_;
      k_dist_conv<<<(int)((n1 + 255) / 256), 256, 0, stream>>>(D0b, 16, 128, F(D + 9), F(D + 6), F(D + 5), 32, AGb);
      k_chstats<<<32, 256, 0, stream>>>(AGb, 32, HW_, MEANb, VARb);
      k_dist_norm<<<(int)((n1 + 255) / 256), 256, 0, stream>>>(AGb, 32, HW_, MEANb, VARb, F(D + 8), F(D + 7));
    }

    // ---- layer norms (eps 1e-5) ----
    k_ln_ch<<<gPix, 256, 0, stream>>>(xin, F(A + A_LN1_G), F(A + A_LN1_B), C_, 1e-5f, XNb, (__bf16*)nullptr);
    k_ln_ch<<<gPix, 256, 0, stream>>>(AGb, F(A + A_LN2_G), F(A + A_LN2_B), AGC_, 1e-5f, AGNb, (__bf16*)nullptr);
    k_ln_ch<<<gPix, 256, 0, stream>>>(win, F(A + A_LN3_G), F(A + A_LN3_B), WC_, 1e-5f, WNb, (__bf16*)nullptr);

    // ---- 1x1 projections, channel-last ----
    k_conv1x1_cl<<<gPix, 256, 0, stream>>>(XNb, C_, WNb, WC_, F(A + A_V_W), F(A + A_V_B), VCL);
    k_conv1x1_cl<<<gPix, 256, 0, stream>>>(XNb, C_, (const float*)nullptr, 0, F(A + A_K_W), F(A + A_K_B), KCL);
    k_conv1x1_cl<<<gPix, 256, 0, stream>>>(WNb, WC_, (const float*)nullptr, 0, F(A + A_Q_W), F(A + A_Q_B), QCL);
    k_conv1x1_cl<<<gPix, 256, 0, stream>>>(AGNb, AGC_, (const float*)nullptr, 0, F(A + A_AG_W), F(A + A_AG_B), ACL);

    // ---- alpha / omega scalars ----
    k_mean_cl<<<B_ * HD_, 256, 0, stream>>>(KCL, KM);
    k_mean_cl<<<B_ * HD_, 256, 0, stream>>>(QCL, QM);
    k_alpha<<<1, 32, 0, stream>>>(KM, QM, F(A + A_CA_W), F(A + A_CA_B), F(A + A_AP_W),
                                  F(A + A_CO_W), F(A + A_CO_B), F(A + A_OP_W), ALv, OMv);

    // ---- fused attention (WMMA) + epilogue ----
    k_attention<<<B_ * 64, 256, 0, stream>>>(KCL, QCL, ACL, VCL, ALv, OMv, MPb);
    k_attn_out<<<gBCHW, 256, 0, stream>>>(MPb, XNb, F(A + A_FAN_W), F(A + A_FAN_B),
                                          F(A + A_GAMMA), F(A + A_DW_W), F(A + A_DW_B), XLAb);

    if (side == 0) {
      // ---- gated CNN (bf16 WMMA GEMMs) ----
      k_ln_ch<<<gPix, 256, 0, stream>>>(XLAb, F(86), F(85), C_, 1e-6f, (float*)nullptr, XLNBF);
      {
        size_t nw1 = (size_t)N1P_ * C_;
        k_wcvt<<<(int)((nw1 + 255) / 256), 256, 0, stream>>>(F(82), N1_, C_, N1P_, C_, W1BF);
        size_t nw2 = (size_t)C_ * K2P_;
        k_wcvt<<<(int)((nw2 + 255) / 256), 256, 0, stream>>>(F(84), C_, HID_, C_, K2P_, W2BF);
      }
      const int tiles1 = ((B_ * HW_) / 16) * (N1P_ / 16); // 2048 * 43
      k_gemm<<<tiles1 / 4, 128, 0, stream>>>(XLNBF, W1BF, F(81), C_, N1P_, N1_, tiles1, Yb);
      k_extract_c<<<gBCHW, 256, 0, stream>>>(Yb, CBUFb);
      k_dwconv7<<<gBCHW, 256, 0, stream>>>(CBUFb, F(80), F(79), CCONVb);
      {
        size_t nu = (size_t)B_ * HW_ * K2P_;
        k_buildU<<<(int)((nu + 255) / 256), 256, 0, stream>>>(Yb, CCONVb, UBF);
      }
      const int tiles2 = ((B_ * HW_) / 16) * (C_ / 16); // 2048 * 8
      k_gemm<<<tiles2 / 4, 128, 0, stream>>>(UBF, W2BF, F(83), K2P_, C_, C_, tiles2, GOb);
      k_final_l<<<gBCHW, 256, 0, stream>>>(xin, XLAb, GOb, F(7), (float*)d_out);
    } else {
      // ---- SE ----
      k_se_reduce<<<B_ * C_, 256, 0, stream>>>(XLAb, YMEANb);
      k_se_gate<<<B_, 128, 0, stream>>>(YMEANb, F(87), F(88), YGb);
      k_final_r<<<gBCHW, 256, 0, stream>>>(xin, XLAb, YGb, F(8),
                                           (float*)d_out + (size_t)B_ * C_ * HW_);
    }
  }
}
